// GAALVExplModule_11089605558297
// MI455X (gfx1250) — compile-verified, matrix operand order checked by reading, not run on hardware
//
#include <hip/hip_runtime.h>
#include <hip/hip_bf16.h>

typedef __attribute__((ext_vector_type(16))) _Float16 v16h;
typedef __attribute__((ext_vector_type(8)))  _Float16 v8h;
typedef __attribute__((ext_vector_type(4)))  _Float16 v4h;
typedef __attribute__((ext_vector_type(8)))  float    v8f;

static constexpr int kNodes   = 50000;
static constexpr int kEdges   = 1600000;
static constexpr int kInFeats = 128;
static constexpr int kEncH    = 20;   // padded to 32 (N) / 64 (K) in WMMA
static constexpr int kDecH    = 64;

// 16-bit WMMA A/B fragment per lane = two contiguous 8-half K-runs:
//   K = kbase..kbase+7 and 16+kbase..16+kbase+7, kbase = 8*(lane>>4),
//   at row (A) / column (B) = lane&15.
__device__ __forceinline__ v8f wmma_f16(v16h a, v16h b, v8f c) {
  return __builtin_amdgcn_wmma_f32_16x16x32_f16(false, a, false, b, (short)0, c,
                                                false, false);
}

__device__ __forceinline__ v16h combine8(v8h lo, v8h hi) {
  v16h r;
#pragma unroll
  for (int i = 0; i < 8; ++i) { r[i] = lo[i]; r[8 + i] = hi[i]; }
  return r;
}

__device__ __forceinline__ void put4(v16h& a, int base, float4 v) {
  a[base + 0] = (_Float16)v.x;
  a[base + 1] = (_Float16)v.y;
  a[base + 2] = (_Float16)v.z;
  a[base + 3] = (_Float16)v.w;
}

// ---------------------------------------------------------------- GCN front
__global__ void k_deg_init(float* __restrict__ deg) {
  int i = blockIdx.x * blockDim.x + threadIdx.x;
  if (i < kNodes) deg[i] = 1.0f;  // self-loop contributes 1 to every in-degree
}

__global__ void k_deg_edges(const long long* __restrict__ eidx,
                            float* __restrict__ deg) {
  int e = blockIdx.x * blockDim.x + threadIdx.x;
  if (e < kEdges) atomicAdd(&deg[eidx[kEdges + e]], 1.0f);
}

// xw = x @ W_gc   (50000x128 @ 128x20) via f16 WMMA, f32 accumulate
__global__ __launch_bounds__(256) void k_xw_wmma(const float* __restrict__ x,
                                                 const float* __restrict__ Wgc,
                                                 float* __restrict__ xw) {
  __shared__ _Float16 WgcT[32][128];  // [n][k], rows 20..31 zero
  // cooperative transposed+padded weight stage (coalesced reads)
  for (int idx = threadIdx.x; idx < 32 * 128; idx += 256)
    (&WgcT[0][0])[idx] = (_Float16)0.f;
  __syncthreads();
  for (int idx = threadIdx.x; idx < kInFeats * kEncH; idx += 256) {
    const int k = idx / kEncH, n = idx % kEncH;
    WgcT[n][k] = (_Float16)Wgc[idx];
  }
  __syncthreads();

  const int lane = threadIdx.x & 31;
  const int wave = threadIdx.x >> 5;
  const int tile = blockIdx.x * 8 + wave;
  const int ntiles = kNodes / 16;  // 3125, exact
  if (tile >= ntiles) return;      // wave-uniform: EXEC stays all-ones
  const int r0 = tile * 16;
  const int lc = lane & 15, lh = lane >> 4;
  const int kbase = 8 * lh;

  v8f c0 = {}, c1 = {};
#pragma unroll
  for (int kc = 0; kc < 4; ++kc) {  // K = 128 -> 4 chunks of 32
    const float* xrow = x + (r0 + lc) * kInFeats + kc * 32 + kbase;
    v16h a;
    put4(a, 0,  ((const float4*)xrow)[0]);
    put4(a, 4,  ((const float4*)xrow)[1]);
    put4(a, 8,  ((const float4*)(xrow + 16))[0]);
    put4(a, 12, ((const float4*)(xrow + 16))[1]);
    const v16h b0 = combine8(*(const v8h*)&WgcT[lc][kc * 32 + kbase],
                             *(const v8h*)&WgcT[lc][kc * 32 + 16 + kbase]);
    const v16h b1 = combine8(*(const v8h*)&WgcT[16 + lc][kc * 32 + kbase],
                             *(const v8h*)&WgcT[16 + lc][kc * 32 + 16 + kbase]);
    c0 = wmma_f16(a, b0, c0);
    c1 = wmma_f16(a, b1, c1);
  }
#pragma unroll
  for (int i = 0; i < 8; ++i) {  // D layout: row = i + 8*lh, col = lc (+16)
    const int row = r0 + i + 8 * lh;
    xw[row * kEncH + lc] = c0[i];
    if (16 + lc < kEncH) xw[row * kEncH + 16 + lc] = c1[i];
  }
}

// dinv = rsqrt(deg); acc starts with the self-loop term xw[n]*dinv[n]
__global__ void k_acc_init(const float* __restrict__ deg,
                           const float* __restrict__ xw,
                           float* __restrict__ dinv, float* __restrict__ acc) {
  int t = blockIdx.x * blockDim.x + threadIdx.x;
  if (t >= kNodes * kEncH) return;
  const int n = t / kEncH, f = t % kEncH;
  const float dv = rsqrtf(deg[n]);  // deg >= 1 always (self loop)
  if (f == 0) dinv[n] = dv;
  acc[t] = xw[t] * dv;
}

// acc[col] += xw[row] * dinv[row]   (final dinv[col] scale applied later)
__global__ void k_scatter(const long long* __restrict__ eidx,
                          const float* __restrict__ xw,
                          const float* __restrict__ dinv,
                          float* __restrict__ acc) {
  int e = blockIdx.x * blockDim.x + threadIdx.x;
  if (e >= kEdges) return;
  const long long r = eidx[e];
  const long long c = eidx[kEdges + e];
  const float dv = dinv[r];
  const float4* xr = reinterpret_cast<const float4*>(xw + r * kEncH);
  float* ac = acc + c * kEncH;
#pragma unroll
  for (int q = 0; q < 5; ++q) {
    float4 v = xr[q];
    atomicAdd(ac + 4 * q + 0, v.x * dv);
    atomicAdd(ac + 4 * q + 1, v.y * dv);
    atomicAdd(ac + 4 * q + 2, v.z * dv);
    atomicAdd(ac + 4 * q + 3, v.w * dv);
  }
}

__global__ void k_x1_final(const float* __restrict__ acc,
                           const float* __restrict__ dinv,
                           const float* __restrict__ bgc,
                           float* __restrict__ x1) {
  int t = blockIdx.x * blockDim.x + threadIdx.x;
  if (t >= kNodes * kEncH) return;
  const int n = t / kEncH, f = t % kEncH;
  x1[t] = fmaxf(dinv[n] * acc[t] + bgc[f], 0.f);
}

// --------------------------------------------- fused per-edge WMMA pipeline
__global__ __launch_bounds__(256) void k_edge_mlp(
    const float* __restrict__ x1, const long long* __restrict__ eidx,
    const int* __restrict__ node_id, const float* __restrict__ noise_n,
    const float* __restrict__ noise_u, const float* __restrict__ Wmu,
    const float* __restrict__ bmu, const float* __restrict__ Wvar,
    const float* __restrict__ bvar, const float* __restrict__ Wd1,
    const float* __restrict__ bd1, const float* __restrict__ Wd2,
    const float* __restrict__ bd2, float* __restrict__ out) {
  __shared__ _Float16 x2buf[8][16][64];  // per-wave: [x1[row]|x1[col]|node|0]
  __shared__ _Float16 zbuf[8][16][32];   // per-wave z round trip (D->A layout)
  __shared__ _Float16 WmuT[32][64];      // [n][k] transposed, zero padded
  __shared__ _Float16 WvaT[32][64];
  __shared__ _Float16 Wd1T[64][32];
  const int lane = threadIdx.x & 31;
  const int wave = threadIdx.x >> 5;
  const int lc = lane & 15, lh = lane >> 4;
  const int kbase = 8 * lh;
  const long long nid = node_id[0];
  const int ntiles = kEdges / 16;  // 100000, exact
  const int wavesTotal = gridDim.x * 8;
  const int wave0 = blockIdx.x * 8 + wave;

  // ---- cooperative transposed weight staging (coalesced global reads)
  for (int idx = threadIdx.x; idx < 32 * 64; idx += 256) {
    (&WmuT[0][0])[idx] = (_Float16)0.f;
    (&WvaT[0][0])[idx] = (_Float16)0.f;
    (&Wd1T[0][0])[idx] = (_Float16)0.f;
  }
  __syncthreads();
  for (int idx = threadIdx.x; idx < 3 * kEncH * kEncH; idx += 256) {
    const int k = idx / kEncH, n = idx % kEncH;   // Wmu/Wvar: [60 x 20]
    WmuT[n][k] = (_Float16)Wmu[idx];
    WvaT[n][k] = (_Float16)Wvar[idx];
  }
  for (int idx = threadIdx.x; idx < kEncH * kDecH; idx += 256) {
    const int k = idx / kDecH, n = idx % kDecH;   // Wd1: [20 x 64]
    Wd1T[n][k] = (_Float16)Wd1[idx];
  }
  __syncthreads();

  // ---- per-wave register B tiles: two aligned 16B LDS reads per tile
  v16h Bmu[2][2], Bva[2][2], Bd1t[4];
#pragma unroll
  for (int kc = 0; kc < 2; ++kc)
#pragma unroll
    for (int nt = 0; nt < 2; ++nt) {
      const int n = nt * 16 + lc;
      Bmu[kc][nt] = combine8(*(const v8h*)&WmuT[n][kc * 32 + kbase],
                             *(const v8h*)&WmuT[n][kc * 32 + 16 + kbase]);
      Bva[kc][nt] = combine8(*(const v8h*)&WvaT[n][kc * 32 + kbase],
                             *(const v8h*)&WvaT[n][kc * 32 + 16 + kbase]);
    }
#pragma unroll
  for (int nt = 0; nt < 4; ++nt)
    Bd1t[nt] = combine8(*(const v8h*)&Wd1T[nt * 16 + lc][kbase],
                        *(const v8h*)&Wd1T[nt * 16 + lc][16 + kbase]);

  // ---- scalar weights/biases (clamped loads + value select, branch-free)
  const int nn1 = 16 + lc;
  const int nn1c = (nn1 < kEncH) ? nn1 : (kEncH - 1);
  float w2v[4], bd1v[4];
#pragma unroll
  for (int nt = 0; nt < 4; ++nt) {
    w2v[nt] = Wd2[nt * 16 + lc];
    bd1v[nt] = bd1[nt * 16 + lc];
  }
  const float bm0 = bmu[lc];
  const float bv0 = bvar[lc];
  const float bm1 = (nn1 < kEncH) ? bmu[nn1c] : 0.f;
  const float bv1 = (nn1 < kEncH) ? bvar[nn1c] : 0.f;
  const float bd2v = bd2[0];

  // ---- loop-invariant x2 columns 40..63: node embedding + zero pad
#pragma unroll
  for (int q = 0; q < 12; ++q) {
    const int cc = 40 + lh * 12 + q;      // 40..51 (lh=0) / 52..63 (lh=1)
    const int qq = cc - 40;               // 0..23
    const int qc = (qq < kEncH) ? qq : (kEncH - 1);
    float v = x1[nid * kEncH + qc];
    v = (qq < kEncH) ? v : 0.f;
    x2buf[wave][lc][cc] = (_Float16)v;
  }

  for (int tile = wave0; tile < ntiles; tile += wavesTotal) {
    const int e0 = tile * 16;
    // cooperative x2 fill: lanes 0-15 bring x1[row_e], lanes 16-31 x1[col_e]
    const long long src = eidx[(lh ? (long long)kEdges : 0ll) + e0 + lc];
    const float* xr = x1 + src * kEncH;
#pragma unroll
    for (int q = 0; q < 5; ++q) {         // 5x float4 load, 5x v4h LDS store
      const float4 v = ((const float4*)xr)[q];
      v4h h4;
      h4[0] = (_Float16)v.x; h4[1] = (_Float16)v.y;
      h4[2] = (_Float16)v.z; h4[3] = (_Float16)v.w;
      *(v4h*)&x2buf[wave][lc][lh * kEncH + 4 * q] = h4;
    }

    // ---- mu / logvar WMMAs (A read back as 2x16B LDS loads per K-chunk);
    //      DS ops are in-order within a wave -> no barrier needed
    v8f cmu0 = {}, cmu1 = {}, cva0 = {}, cva1 = {};
#pragma unroll
    for (int kc = 0; kc < 2; ++kc) {
      const v8h lo = *(const v8h*)&x2buf[wave][lc][kc * 32 + kbase];
      const v8h hi = *(const v8h*)&x2buf[wave][lc][kc * 32 + 16 + kbase];
      const v16h a = combine8(lo, hi);
      cmu0 = wmma_f16(a, Bmu[kc][0], cmu0);
      cmu1 = wmma_f16(a, Bmu[kc][1], cmu1);
      cva0 = wmma_f16(a, Bva[kc][0], cva0);
      cva1 = wmma_f16(a, Bva[kc][1], cva1);
    }

    // ---- z = mu + exp(logvar)*noise (branch-free clamped noise loads)
    const int f1 = 16 + lc;
    const int f1c = (f1 < kEncH) ? f1 : (kEncH - 1);
#pragma unroll
    for (int i = 0; i < 8; ++i) {
      const int el = i + 8 * lh;          // D-layout row = local edge
      const long long ee = e0 + el;
      const float n0 = noise_n[ee * kEncH + lc];
      const float n1 = noise_n[ee * kEncH + f1c];
      const float z0 = (cmu0[i] + bm0) + __expf(cva0[i] + bv0) * n0;
      float z1 = (cmu1[i] + bm1) + __expf(cva1[i] + bv1) * n1;
      z1 = (f1 < kEncH) ? z1 : 0.f;       // pads feats 20..31 with zero
      zbuf[wave][el][lc] = (_Float16)z0;
      zbuf[wave][el][f1] = (_Float16)z1;
    }

    // ---- h = relu(z @ W_d1 + b_d1): A from LDS, N = 64 over 4 WMMAs
    const v16h az = combine8(*(const v8h*)&zbuf[wave][lc][kbase],
                             *(const v8h*)&zbuf[wave][lc][16 + kbase]);
    v8f ch0 = {}, ch1 = {}, ch2 = {}, ch3 = {};
    ch0 = wmma_f16(az, Bd1t[0], ch0);
    ch1 = wmma_f16(az, Bd1t[1], ch1);
    ch2 = wmma_f16(az, Bd1t[2], ch2);
    ch3 = wmma_f16(az, Bd1t[3], ch3);

    // ---- sw = relu(h @ W_d2 + b_d2): per-lane partials + 16-lane reduce
    float partial[8];
#pragma unroll
    for (int i = 0; i < 8; ++i) {
      float s = fmaxf(ch0[i] + bd1v[0], 0.f) * w2v[0];
      s += fmaxf(ch1[i] + bd1v[1], 0.f) * w2v[1];
      s += fmaxf(ch2[i] + bd1v[2], 0.f) * w2v[2];
      s += fmaxf(ch3[i] + bd1v[3], 0.f) * w2v[3];
      partial[i] = s;
    }
#pragma unroll
    for (int i = 0; i < 8; ++i) {
      partial[i] += __shfl_xor(partial[i], 1, 32);
      partial[i] += __shfl_xor(partial[i], 2, 32);
      partial[i] += __shfl_xor(partial[i], 4, 32);
      partial[i] += __shfl_xor(partial[i], 8, 32);
    }

    // ---- gumbel-sigmoid tail: lanes 0 (edges 0-7) and 16 (edges 8-15)
    if (lc == 0) {
#pragma unroll
      for (int i = 0; i < 8; ++i) {
        const long long ee = e0 + 8 * lh + i;
        const float sw = fmaxf(partial[i] + bd2v, 0.f);
        const float u = noise_u[ee];
        const float kB = 1e-4f;
        const float ev = (kB - (1.f - kB)) * u + (1.f - kB);
        const float gate = __logf(ev) - __logf(1.f - ev) + sw;
        out[ee] = 1.f / (1.f + __expf(-gate));
      }
    }
  }
}

// --------------------------------------------------------------- dispatcher
extern "C" void kernel_launch(void* const* d_in, const int* in_sizes, int n_in,
                              void* d_out, int out_size, void* d_ws,
                              size_t ws_size, hipStream_t stream) {
  const float* x        = (const float*)d_in[0];
  const long long* eidx = (const long long*)d_in[1];  // int64 per reference
  const int* node_id    = (const int*)d_in[2];
  const float* noise_n  = (const float*)d_in[3];
  const float* noise_u  = (const float*)d_in[4];
  const float* Wgc      = (const float*)d_in[5];
  const float* bgc      = (const float*)d_in[6];
  const float* Wmu      = (const float*)d_in[7];
  const float* bmu      = (const float*)d_in[8];
  const float* Wvar     = (const float*)d_in[9];
  const float* bvar     = (const float*)d_in[10];
  const float* Wd1      = (const float*)d_in[11];
  const float* bd1      = (const float*)d_in[12];
  const float* Wd2      = (const float*)d_in[13];
  const float* bd2      = (const float*)d_in[14];
  float* out = (float*)d_out;

  float* ws   = (float*)d_ws;          // ~12.4 MB of f32 scratch
  float* xw   = ws;                    // [50000 x 20]
  float* deg  = ws + 1000000;          // [50000]
  float* dinv = ws + 1050000;          // [50000]
  float* acc  = ws + 1100000;          // [50000 x 20]
  float* x1   = ws + 2100000;          // [50000 x 20]

  k_deg_init<<<(kNodes + 255) / 256, 256, 0, stream>>>(deg);
  k_deg_edges<<<(kEdges + 255) / 256, 256, 0, stream>>>(eidx, deg);
  k_xw_wmma<<<(kNodes / 16 + 7) / 8, 256, 0, stream>>>(x, Wgc, xw);
  k_acc_init<<<(kNodes * kEncH + 255) / 256, 256, 0, stream>>>(deg, xw, dinv, acc);
  k_scatter<<<(kEdges + 255) / 256, 256, 0, stream>>>(eidx, xw, dinv, acc);
  k_x1_final<<<(kNodes * kEncH + 255) / 256, 256, 0, stream>>>(acc, dinv, bgc, x1);
  k_edge_mlp<<<1024, 256, 0, stream>>>(x1, eidx, node_id, noise_n, noise_u, Wmu,
                                       bmu, Wvar, bvar, Wd1, bd1, Wd2, bd2, out);
}
